// NaiveRNN_39127152066628
// MI455X (gfx1250) — compile-verified, matrix operand order checked by reading, not run on hardware
//
#include <hip/hip_runtime.h>

// ---------------------------------------------------------------------------
// CDNA5 (gfx1250) ReLU-RNN:
//   Phase 1: x_proj = X @ W_ih^T + b_ih   (one big bf16 WMMA GEMM -> d_out)
//   Phase 2: 2048x   h = relu(xp_t + h @ W_hh^T + b_hh)  (per-step WMMA GEMM,
//            bf16 h ping-pong in d_ws, d_out updated in place)
// ---------------------------------------------------------------------------

typedef __attribute__((ext_vector_type(16))) __bf16 v16bf;
typedef __attribute__((ext_vector_type(8)))  float  v8f;

union FragU {
    uint4          q[2];
    unsigned short s[16];
    v16bf          v;
};

// Native hardware bf16 convert (RNE) — lowers to packed cvt ops on gfx1250,
// replacing the 3-VALU-per-element integer emulation from the last round.
__device__ __forceinline__ unsigned short f32_to_bf16_hw(float f) {
    __bf16 b = (__bf16)f;
    return __builtin_bit_cast(unsigned short, b);
}

// bf16 fragment: lane holds 8 contiguous K at +0 and 8 contiguous K at +16
// (per ISA 16-bit A/B 16x32 VGPR layout). `base` already points at
// row*ld + k0 + kbase.
__device__ __forceinline__ v16bf load_frag_bf16(const unsigned short* base) {
    FragU f;
    f.q[0] = *(const uint4*)(base);        // K kb .. kb+7
    f.q[1] = *(const uint4*)(base + 16);   // K kb+16 .. kb+23
    return f.v;
}

// Same fragment but source is fp32: load 16 floats, convert (hardware RNE).
__device__ __forceinline__ v16bf load_frag_f32(const float* base) {
    const float4* p0 = (const float4*)(base);
    const float4* p1 = (const float4*)(base + 16);
    float4 a = p0[0], b = p0[1];
    float4 c = p1[0], d = p1[1];
    FragU f;
    f.v[0]  = (__bf16)a.x; f.v[1]  = (__bf16)a.y;
    f.v[2]  = (__bf16)a.z; f.v[3]  = (__bf16)a.w;
    f.v[4]  = (__bf16)b.x; f.v[5]  = (__bf16)b.y;
    f.v[6]  = (__bf16)b.z; f.v[7]  = (__bf16)b.w;
    f.v[8]  = (__bf16)c.x; f.v[9]  = (__bf16)c.y;
    f.v[10] = (__bf16)c.z; f.v[11] = (__bf16)c.w;
    f.v[12] = (__bf16)d.x; f.v[13] = (__bf16)d.y;
    f.v[14] = (__bf16)d.z; f.v[15] = (__bf16)d.w;
    return f.v;
}

// ---------------------------------------------------------------------------
// small utility kernels
// ---------------------------------------------------------------------------
__global__ void cvt_f32_to_bf16_kernel(const float* __restrict__ in,
                                       unsigned short* __restrict__ out, int n) {
    int i = blockIdx.x * blockDim.x + threadIdx.x;
    if (i < n) out[i] = f32_to_bf16_hw(in[i]);
}

__global__ void zero_bf16_kernel(unsigned short* __restrict__ p, int n) {
    int i = blockIdx.x * blockDim.x + threadIdx.x;
    if (i < n) p[i] = 0;
}

// ---------------------------------------------------------------------------
// Phase 1: C[M,H] = X[M,I] * Wih[H,I]^T + b_ih   (M = B*T)
// One wave computes a 16(M) x 64(N) strip; block = 8 waves covering 16 x 512;
// 2 blocks per M-tile. A (fp32) converted on the fly; B pre-converted bf16.
// ---------------------------------------------------------------------------
__global__ __launch_bounds__(256) void xproj_gemm_kernel(
    const float* __restrict__ X,            // [M, I] fp32
    const unsigned short* __restrict__ Wb,  // [H, I] bf16
    const float* __restrict__ bias,         // [H]
    float* __restrict__ out,                // [M, H]
    int I, int H)
{
    const int lane  = threadIdx.x & 31;
    const int wave  = threadIdx.x >> 5;
    const int mtile = blockIdx.x >> 1;
    const int nhalf = blockIdx.x & 1;
    const int m0    = mtile * 16;
    const int n0    = nhalf * 512 + wave * 64;

    const int row = lane & 15;
    const int kb  = (lane >> 4) * 8;

    const float* aRow = X + (size_t)(m0 + row) * I + kb;
    const unsigned short* bRow0 = Wb + (size_t)(n0 +  0 + row) * I + kb;
    const unsigned short* bRow1 = Wb + (size_t)(n0 + 16 + row) * I + kb;
    const unsigned short* bRow2 = Wb + (size_t)(n0 + 32 + row) * I + kb;
    const unsigned short* bRow3 = Wb + (size_t)(n0 + 48 + row) * I + kb;

    v8f acc[4] = {v8f{}, v8f{}, v8f{}, v8f{}};

    for (int k = 0; k < I; k += 32) {
        v16bf a  = load_frag_f32(aRow + k);
        v16bf b0 = load_frag_bf16(bRow0 + k);
        acc[0] = __builtin_amdgcn_wmma_f32_16x16x32_bf16(false, a, false, b0,
                                                         (short)0, acc[0], false, false);
        v16bf b1 = load_frag_bf16(bRow1 + k);
        acc[1] = __builtin_amdgcn_wmma_f32_16x16x32_bf16(false, a, false, b1,
                                                         (short)0, acc[1], false, false);
        v16bf b2 = load_frag_bf16(bRow2 + k);
        acc[2] = __builtin_amdgcn_wmma_f32_16x16x32_bf16(false, a, false, b2,
                                                         (short)0, acc[2], false, false);
        v16bf b3 = load_frag_bf16(bRow3 + k);
        acc[3] = __builtin_amdgcn_wmma_f32_16x16x32_bf16(false, a, false, b3,
                                                         (short)0, acc[3], false, false);
    }

    // C/D layout: VGPR r -> M = r + (lane>=16 ? 8 : 0), N = lane & 15
    const int ncol = lane & 15;
    const int mofs = m0 + (lane >> 4) * 8;
#pragma unroll
    for (int j = 0; j < 4; ++j) {
        const int n = n0 + j * 16 + ncol;
        const float bj = bias[n];
#pragma unroll
        for (int r = 0; r < 8; ++r) {
            out[(size_t)(mofs + r) * H + n] = acc[j][r] + bj;
        }
    }
}

// ---------------------------------------------------------------------------
// Phase 2: one timestep.  h_new = relu(xp_t + h_prev @ Whh^T + b_hh)
// One wave per 16x16 tile: (B/16)*(H/16) = 256 waves = 32 blocks of 8 waves.
// d_out row for batch b at time t holds xp on entry, h_new on exit (in place).
// ---------------------------------------------------------------------------
__global__ __launch_bounds__(256) void rnn_step_kernel(
    const unsigned short* __restrict__ hprev, // [B, H] bf16
    const unsigned short* __restrict__ Whh,   // [H, H] bf16
    const float* __restrict__ bhh,            // [H]
    float* __restrict__ out,                  // [B, T, H] fp32 (in-place xp -> h)
    unsigned short* __restrict__ hnext,       // [B, H] bf16
    int t, int T, int H)
{
    const int lane  = threadIdx.x & 31;
    const int wid   = blockIdx.x * (blockDim.x >> 5) + (threadIdx.x >> 5);
    const int ntlN  = H >> 4;             // 64 N-tiles
    const int mtile = wid / ntlN;
    const int ntile = wid % ntlN;
    const int m0 = mtile * 16;
    const int n0 = ntile * 16;

    const int row = lane & 15;
    const int kb  = (lane >> 4) * 8;

    const unsigned short* aRow = hprev + (size_t)(m0 + row) * H + kb;
    const unsigned short* bRow = Whh   + (size_t)(n0 + row) * H + kb;

    v8f acc = {};
#pragma unroll 4
    for (int k = 0; k < H; k += 32) {
        v16bf a = load_frag_bf16(aRow + k);
        v16bf b = load_frag_bf16(bRow + k);
        acc = __builtin_amdgcn_wmma_f32_16x16x32_bf16(false, a, false, b,
                                                      (short)0, acc, false, false);
    }

    const int n    = n0 + (lane & 15);
    const int mofs = m0 + (lane >> 4) * 8;
    const float bj = bhh[n];
#pragma unroll
    for (int r = 0; r < 8; ++r) {
        const int b_idx = mofs + r;
        const size_t o  = ((size_t)b_idx * T + t) * H + n;
        float v = acc[r] + out[o] + bj;
        v = v > 0.0f ? v : 0.0f;
        out[o] = v;
        hnext[(size_t)b_idx * H + n] = f32_to_bf16_hw(v);
    }
}

// ---------------------------------------------------------------------------
// host driver
// ---------------------------------------------------------------------------
extern "C" void kernel_launch(void* const* d_in, const int* in_sizes, int n_in,
                              void* d_out, int out_size, void* d_ws, size_t ws_size,
                              hipStream_t stream) {
    (void)in_sizes; (void)n_in; (void)out_size; (void)ws_size;

    const float* X   = (const float*)d_in[0];  // [B, T, I]
    const float* Wih = (const float*)d_in[1];  // [H, I]
    const float* bih = (const float*)d_in[2];  // [H]
    const float* Whh = (const float*)d_in[3];  // [H, H]
    const float* bhh = (const float*)d_in[4];  // [H]
    float* out = (float*)d_out;                // [B, T, H]

    const int B = 64, T = 2048, I = 512, H = 1024;
    const int M = B * T;

    // workspace layout (bf16 as ushort): ~3.3 MB total
    unsigned short* whh_bf = (unsigned short*)d_ws;            // H*H   = 2 MB
    unsigned short* wih_bf = whh_bf + (size_t)H * H;           // H*I   = 1 MB
    unsigned short* hbuf0  = wih_bf + (size_t)H * I;           // B*H   = 128 KB
    unsigned short* hbuf1  = hbuf0 + (size_t)B * H;            // B*H   = 128 KB

    // one-time (per call) conversions + h0 = 0
    cvt_f32_to_bf16_kernel<<<(H * H + 255) / 256, 256, 0, stream>>>(Whh, whh_bf, H * H);
    cvt_f32_to_bf16_kernel<<<(H * I + 255) / 256, 256, 0, stream>>>(Wih, wih_bf, H * I);
    zero_bf16_kernel<<<(B * H + 255) / 256, 256, 0, stream>>>(hbuf0, B * H);

    // Phase 1: x_proj into d_out.  (M/16) M-tiles, 2 blocks per tile.
    xproj_gemm_kernel<<<(M / 16) * 2, 256, 0, stream>>>(X, wih_bf, bih, out, I, H);

    // Phase 2: sequential scan, one small WMMA GEMM per timestep.
    const int step_blocks = ((B / 16) * (H / 16)) / 8;   // 256 waves / 8 per block = 32
    for (int t = 0; t < T; ++t) {
        const unsigned short* hp = (t & 1) ? hbuf1 : hbuf0;
        unsigned short*       hn = (t & 1) ? hbuf0 : hbuf1;
        rnn_step_kernel<<<step_blocks, 256, 0, stream>>>(hp, whh_bf, bhh, out, hn, t, T, H);
    }
}
